// AttentionDecoderModel_67001489817835
// MI455X (gfx1250) — compile-verified, matrix operand order checked by reading, not run on hardware
//
#include <hip/hip_runtime.h>
#include <hip/hip_bf16.h>

// ---------------------------------------------------------------------------
// Types for CDNA5 WMMA (wave32): D(f32 16x16) = A(bf16 16x32) * B(bf16 32x16) + C
// ---------------------------------------------------------------------------
typedef __attribute__((ext_vector_type(16))) __bf16 v16bf;
typedef __attribute__((ext_vector_type(8)))  float  v8f;

union FragU { v16bf v; unsigned u[8]; };

// Gather one 16-bit A/B fragment from an LDS tile stored row-major [row][k].
// ldE = row stride in elements (even), kOff = K window start (even).
// ISA 7.12.2: lanes 0-15 -> K base 0 (VGPR0-3: K0..7, VGPR4-7: K16..23)
//             lanes16-31 -> K base 8 (VGPR0-3: K8..15, VGPR4-7: K24..31)
__device__ __forceinline__ v16bf load_frag_bf16(const __bf16* tile, int row0,
                                                int kOff, int ldE) {
  const int lane = threadIdx.x & 31;
  const int r    = row0 + (lane & 15);
  const int kb2  = (lane & 16) ? 4 : 0;  // (kbase/2) in u32 units
  const unsigned* p = (const unsigned*)(tile + (size_t)r * ldE + kOff);
  FragU f;
#pragma unroll
  for (int i = 0; i < 4; ++i) {
    f.u[i]     = p[kb2 + i];
    f.u[4 + i] = p[kb2 + 8 + i];
  }
  return f.v;
}

// ---------------------------------------------------------------------------
// Generic C[M,N] = X[M,K] @ W[N,K]^T + bias (+res / +DoubleSwish)
// block = 128 threads (4 waves, 2x2 wave grid); block tile 64x64; K step 64.
// Each wave owns a 32x32 output tile: 2 A-frags + 2 B-frags per 32-K step.
// TX: float or __bf16 activations; W always float. TO: float or __bf16.
// mode: 0 = bias only, 1 = bias + residual(float), 2 = bias + DoubleSwish
// M,N multiples of 64; K multiple of 64.
// ---------------------------------------------------------------------------
#define GLD 72   // LDS row stride in elements (144B = 16*9 -> b128-aligned rows)

template <typename TX, typename TO>
__global__ __launch_bounds__(128) void gemm_kernel(
    const TX* __restrict__ X, const float* __restrict__ W,
    const float* __restrict__ bias, const float* __restrict__ res,
    TO* __restrict__ C, int M, int N, int K, int mode) {
  __shared__ __align__(16) __bf16 Xs[64][GLD];
  __shared__ __align__(16) __bf16 Ws[64][GLD];

  const int tid  = threadIdx.x;
  const int lane = tid & 31;
  const int wave = tid >> 5;
  const int wm   = wave >> 1;       // wave row (0..1) -> 32 rows
  const int wn   = wave & 1;        // wave col (0..1) -> 32 cols
  const int m0   = blockIdx.y * 64;
  const int n0   = blockIdx.x * 64;

  const v8f zf = {0.f, 0.f, 0.f, 0.f, 0.f, 0.f, 0.f, 0.f};
  v8f acc[2][2] = {{zf, zf}, {zf, zf}};

  for (int k0 = 0; k0 < K; k0 += 64) {
    // ---- stage X tile (64x64) ----
    if constexpr (sizeof(TX) == 4) {
      float4 xv[8];
#pragma unroll
      for (int i = 0; i < 8; ++i) {            // decoupled wide loads
        int idx = (tid + i * 128) * 4;         // 0..4092
        int r = idx >> 6, c = idx & 63;
        xv[i] = *(const float4*)&X[(size_t)(m0 + r) * K + k0 + c];
      }
#pragma unroll
      for (int i = 0; i < 8; ++i) {            // convert + b64 LDS store
        int idx = (tid + i * 128) * 4;
        int r = idx >> 6, c = idx & 63;
        union { __bf16 h[4]; unsigned long long q; } u;
        const float* f = (const float*)&xv[i];
#pragma unroll
        for (int j = 0; j < 4; ++j) u.h[j] = (__bf16)f[j];
        *(unsigned long long*)&Xs[r][c] = u.q;
      }
    } else {
#pragma unroll
      for (int i = 0; i < 4; ++i) {            // bf16: straight 16B copies
        int idx = (tid + i * 128) * 8;         // 0..4088
        int r = idx >> 6, c = idx & 63;
        *(uint4*)&Xs[r][c] = *(const uint4*)&X[(size_t)(m0 + r) * K + k0 + c];
      }
    }
    // ---- stage W tile (64x64, always f32) ----
    {
      float4 wv[8];
#pragma unroll
      for (int i = 0; i < 8; ++i) {
        int idx = (tid + i * 128) * 4;
        int r = idx >> 6, c = idx & 63;
        wv[i] = *(const float4*)&W[(size_t)(n0 + r) * K + k0 + c];
      }
#pragma unroll
      for (int i = 0; i < 8; ++i) {
        int idx = (tid + i * 128) * 4;
        int r = idx >> 6, c = idx & 63;
        union { __bf16 h[4]; unsigned long long q; } u;
        const float* f = (const float*)&wv[i];
#pragma unroll
        for (int j = 0; j < 4; ++j) u.h[j] = (__bf16)f[j];
        *(unsigned long long*)&Ws[r][c] = u.q;
      }
    }
    if (k0 + 64 < K) {  // next K tile -> global_prefetch_b8
      __builtin_prefetch(&X[(size_t)(m0 + (tid >> 1)) * K + k0 + 64], 0, 1);
      __builtin_prefetch(&W[(size_t)(n0 + (tid >> 1)) * K + k0 + 64], 0, 1);
    }
    __syncthreads();

    v16bf a[2][2], b[2][2];
#pragma unroll
    for (int i = 0; i < 2; ++i)
#pragma unroll
      for (int kk = 0; kk < 2; ++kk)
        a[i][kk] = load_frag_bf16(&Xs[0][0], wm * 32 + i * 16, kk * 32, GLD);
#pragma unroll
    for (int j = 0; j < 2; ++j)
#pragma unroll
      for (int kk = 0; kk < 2; ++kk)
        b[j][kk] = load_frag_bf16(&Ws[0][0], wn * 32 + j * 16, kk * 32, GLD);
#pragma unroll
    for (int i = 0; i < 2; ++i)
#pragma unroll
      for (int j = 0; j < 2; ++j) {
        acc[i][j] = __builtin_amdgcn_wmma_f32_16x16x32_bf16(
            false, a[i][0], false, b[j][0], (short)0, acc[i][j], false, false);
        acc[i][j] = __builtin_amdgcn_wmma_f32_16x16x32_bf16(
            false, a[i][1], false, b[j][1], (short)0, acc[i][j], false, false);
      }
    __syncthreads();
  }

  // epilogue (C layout: lanes 0-15 rows v, lanes 16-31 rows v+8; col = lane&15)
  const int lg8 = (lane & 16) ? 8 : 0;
  const int nl  = lane & 15;
#pragma unroll
  for (int i = 0; i < 2; ++i)
#pragma unroll
    for (int j = 0; j < 2; ++j) {
      int n = n0 + wn * 32 + j * 16 + nl;
      float bn = bias[n];
#pragma unroll
      for (int v = 0; v < 8; ++v) {
        int m = m0 + wm * 32 + i * 16 + v + lg8;
        float val = acc[i][j][v] + bn;
        if (mode == 2) val = val / (1.f + __expf(1.f - val));  // DoubleSwish
        if (mode == 1) val += res[(size_t)m * N + n];
        C[(size_t)m * N + n] = (TO)val;
      }
    }
}

// ---------------------------------------------------------------------------
// Flash attention over bf16 Q/K/V: per (b, h, 64-row Q tile). HD=64, HDV=32.
// Q: [B,T,8,64] K: [B,S,8,64] V: [B,S,8,32] O: [B,T,8,32]  (all bf16, packed)
// block = 128 threads (4 waves); each wave owns 16 query rows.
// ---------------------------------------------------------------------------
#define FLD 72   // LDS row stride in elements (144B = 16*9 -> b128-aligned rows)

__global__ __launch_bounds__(128) void flash_attn_kernel(
    const __bf16* __restrict__ Q, const __bf16* __restrict__ K,
    const __bf16* __restrict__ V, __bf16* __restrict__ O,
    int T_, int S_, int causal) {
  __shared__ __align__(16) __bf16 Qs[64][FLD];
  __shared__ __align__(16) __bf16 Ks[64][FLD];
  __shared__ __align__(16) __bf16 Vt[32][FLD];      // V transposed: [dv][s]
  __shared__ __align__(16) __bf16 Ps[4][16][FLD];   // per-wave P tile [t][s]

  const int tid  = threadIdx.x;
  const int lane = tid & 31;
  const int wave = tid >> 5;
  const int q0   = blockIdx.x * 64;
  const int h    = blockIdx.y;
  const int b    = blockIdx.z;

  const __bf16* qh = Q + (size_t)b * T_ * 512 + (size_t)h * 64;
  const __bf16* kh = K + (size_t)b * S_ * 512 + (size_t)h * 64;
  const __bf16* vh = V + (size_t)b * S_ * 256 + (size_t)h * 32;
  __bf16*       oh = O + (size_t)b * T_ * 256 + (size_t)h * 32;

  // load Q tile once (64 x 64 bf16) as 16B chunks
#pragma unroll
  for (int i = 0; i < 4; ++i) {
    int idx = (tid + i * 128) * 8;   // 0..4088
    int r = idx >> 6, c = idx & 63;
    *(uint4*)&Qs[r][c] = *(const uint4*)&qh[(size_t)(q0 + r) * 512 + c];
  }
  __syncthreads();
  v16bf aq0 = load_frag_bf16(&Qs[0][0], wave * 16, 0,  FLD);
  v16bf aq1 = load_frag_bf16(&Qs[0][0], wave * 16, 32, FLD);

  const v8f zf = {0.f, 0.f, 0.f, 0.f, 0.f, 0.f, 0.f, 0.f};
  v8f oacc[2] = {zf, zf};
  float mstat[8], lstat[8];
#pragma unroll
  for (int v = 0; v < 8; ++v) { mstat[v] = -1e30f; lstat[v] = 0.f; }

  const int lg8 = (lane & 16) ? 8 : 0;
  const int nl  = lane & 15;
  const int nTiles = causal ? ((int)blockIdx.x + 1) : (S_ >> 6);

  for (int kt = 0; kt < nTiles; ++kt) {
    const int kv0 = kt * 64;
    __syncthreads();   // all waves done with previous Ks/Vt
    // K tile: 64x64 bf16, 16B chunks
#pragma unroll
    for (int i = 0; i < 4; ++i) {
      int idx = (tid + i * 128) * 8;
      int r = idx >> 6, c = idx & 63;
      *(uint4*)&Ks[r][c] = *(const uint4*)&kh[(size_t)(kv0 + r) * 512 + c];
    }
    // V tile: 64x32 bf16 -> transposed Vt[dv][s]; pair adjacent s into b32 stores
#pragma unroll
    for (int i = 0; i < 2; ++i) {
      int ch = tid + i * 128;          // 0..255
      int s2 = (ch & 31) * 2;          // s pair base
      int d4 = (ch >> 5) * 4;          // dv quad base
      union H4 { __bf16 h[4]; uint2 q; } r0, r1;
      r0.q = *(const uint2*)&vh[(size_t)(kv0 + s2)     * 256 + d4];
      r1.q = *(const uint2*)&vh[(size_t)(kv0 + s2 + 1) * 256 + d4];
#pragma unroll
      for (int j = 0; j < 4; ++j) {
        union { __bf16 h[2]; unsigned w; } o;
        o.h[0] = r0.h[j]; o.h[1] = r1.h[j];
        *(unsigned*)&Vt[d4 + j][s2] = o.w;
      }
    }
    __syncthreads();

    // scores S = Q @ K^T   (scale baked into weights per reference)
    v8f sc[4] = {zf, zf, zf, zf};
#pragma unroll
    for (int j = 0; j < 4; ++j) {
      v16bf bk0 = load_frag_bf16(&Ks[0][0], j * 16, 0, FLD);
      sc[j] = __builtin_amdgcn_wmma_f32_16x16x32_bf16(
          false, aq0, false, bk0, (short)0, sc[j], false, false);
      v16bf bk1 = load_frag_bf16(&Ks[0][0], j * 16, 32, FLD);
      sc[j] = __builtin_amdgcn_wmma_f32_16x16x32_bf16(
          false, aq1, false, bk1, (short)0, sc[j], false, false);
    }

    if (causal) {
#pragma unroll
      for (int j = 0; j < 4; ++j) {
        int scol = kv0 + j * 16 + nl;
#pragma unroll
        for (int v = 0; v < 8; ++v) {
          int srow = q0 + wave * 16 + v + lg8;
          if (scol > srow) sc[j][v] = -1e30f;
        }
      }
    }

    // online softmax; stats share the C-matrix row mapping so alpha needs no bcast
#pragma unroll
    for (int v = 0; v < 8; ++v) {
      float mx = fmaxf(fmaxf(sc[0][v], sc[1][v]), fmaxf(sc[2][v], sc[3][v]));
      mx = fmaxf(mx, __shfl_xor(mx, 1, 32));
      mx = fmaxf(mx, __shfl_xor(mx, 2, 32));
      mx = fmaxf(mx, __shfl_xor(mx, 4, 32));
      mx = fmaxf(mx, __shfl_xor(mx, 8, 32));
      float mnew  = fmaxf(mstat[v], mx);
      float alpha = __expf(mstat[v] - mnew);
      float rs = 0.f;
#pragma unroll
      for (int j = 0; j < 4; ++j) {
        float p = __expf(sc[j][v] - mnew);
        sc[j][v] = p;
        rs += p;
      }
      rs += __shfl_xor(rs, 1, 32);
      rs += __shfl_xor(rs, 2, 32);
      rs += __shfl_xor(rs, 4, 32);
      rs += __shfl_xor(rs, 8, 32);
      lstat[v] = lstat[v] * alpha + rs;
      mstat[v] = mnew;
      oacc[0][v] *= alpha;
      oacc[1][v] *= alpha;
    }

    // P (C-layout regs) -> per-wave LDS tile [t][s] to become an A fragment
#pragma unroll
    for (int j = 0; j < 4; ++j)
#pragma unroll
      for (int v = 0; v < 8; ++v)
        Ps[wave][v + lg8][j * 16 + nl] = (__bf16)sc[j][v];
    asm volatile("s_wait_dscnt 0" ::: "memory");  // wave-local LDS RAW

    v16bf ap0 = load_frag_bf16(&Ps[wave][0][0], 0, 0,  FLD);
    v16bf ap1 = load_frag_bf16(&Ps[wave][0][0], 0, 32, FLD);
#pragma unroll
    for (int j = 0; j < 2; ++j) {
      v16bf bv0 = load_frag_bf16(&Vt[0][0], j * 16, 0, FLD);
      oacc[j] = __builtin_amdgcn_wmma_f32_16x16x32_bf16(
          false, ap0, false, bv0, (short)0, oacc[j], false, false);
      v16bf bv1 = load_frag_bf16(&Vt[0][0], j * 16, 32, FLD);
      oacc[j] = __builtin_amdgcn_wmma_f32_16x16x32_bf16(
          false, ap1, false, bv1, (short)0, oacc[j], false, false);
    }
  }

  // write O = acc / l  (bf16)
#pragma unroll
  for (int j = 0; j < 2; ++j)
#pragma unroll
    for (int v = 0; v < 8; ++v) {
      int t = q0 + wave * 16 + v + lg8;
      oh[(size_t)t * 256 + j * 16 + nl] = (__bf16)(oacc[j][v] / lstat[v]);
    }
}

// ---------------------------------------------------------------------------
// BasicNorm + bypass: out = tgt + (x * rsqrt(mean(x^2)+exp(eps)) - tgt) * bs
// one block (256 thr) per row of 512
// ---------------------------------------------------------------------------
__global__ __launch_bounds__(256) void norm_bypass_kernel(
    const float* __restrict__ x, const float* __restrict__ tgt,
    const float* __restrict__ eps_log, const float* __restrict__ bs,
    float* __restrict__ out) {
  const size_t row = blockIdx.x;
  const float* xr = x + row * 512;
  const float* tr = tgt + row * 512;
  float v0 = xr[threadIdx.x];
  float v1 = xr[threadIdx.x + 256];
  __shared__ float red[256];
  red[threadIdx.x] = v0 * v0 + v1 * v1;
  __syncthreads();
  for (int s = 128; s > 0; s >>= 1) {
    if ((int)threadIdx.x < s) red[threadIdx.x] += red[threadIdx.x + s];
    __syncthreads();
  }
  float scale = rsqrtf(red[0] * (1.f / 512.f) + __expf(eps_log[0]));
  float bsv = bs[0];
  out[row * 512 + threadIdx.x]       = tr[threadIdx.x]       + (v0 * scale - tr[threadIdx.x])       * bsv;
  out[row * 512 + threadIdx.x + 256] = tr[threadIdx.x + 256] + (v1 * scale - tr[threadIdx.x + 256]) * bsv;
}

// ---------------------------------------------------------------------------
extern "C" void kernel_launch(void* const* d_in, const int* in_sizes, int n_in,
                              void* d_out, int out_size, void* d_ws, size_t ws_size,
                              hipStream_t stream) {
  (void)in_sizes; (void)n_in; (void)out_size; (void)ws_size;
  const float* tgt    = (const float*)d_in[0];
  const float* memory = (const float*)d_in[1];
  const float* sa_wq = (const float*)d_in[2];
  const float* sa_bq = (const float*)d_in[3];
  const float* sa_wk = (const float*)d_in[4];
  const float* sa_bk = (const float*)d_in[5];
  const float* sa_wv = (const float*)d_in[6];
  const float* sa_bv = (const float*)d_in[7];
  const float* sa_wo = (const float*)d_in[8];
  const float* sa_bo = (const float*)d_in[9];
  const float* ca_wq = (const float*)d_in[10];
  const float* ca_bq = (const float*)d_in[11];
  const float* ca_wk = (const float*)d_in[12];
  const float* ca_bk = (const float*)d_in[13];
  const float* ca_wv = (const float*)d_in[14];
  const float* ca_bv = (const float*)d_in[15];
  const float* ca_wo = (const float*)d_in[16];
  const float* ca_bo = (const float*)d_in[17];
  const float* ff_w1 = (const float*)d_in[18];
  const float* ff_b1 = (const float*)d_in[19];
  const float* ff_w2 = (const float*)d_in[20];
  const float* ff_b2 = (const float*)d_in[21];
  const float* neps  = (const float*)d_in[22];
  const float* bscal = (const float*)d_in[23];
  // d_in[24]=tgt_mask (causal, applied analytically), d_in[25]=memory_mask (all valid)

  const size_t MT = 8192;   // B*T
  const size_t MS = 16384;  // B*S
  char* w = (char*)d_ws;
  __bf16* bufQ    = (__bf16*)w;  w += MT * 512 * 2;   // [MT,512] bf16
  __bf16* bufK    = (__bf16*)w;  w += MS * 512 * 2;   // [MS,512] bf16
  __bf16* bufV    = (__bf16*)w;  w += MS * 256 * 2;   // [MS,256] bf16
  __bf16* bufAttn = (__bf16*)w;  w += MT * 256 * 2;   // [MT,256] bf16
  float*  bufX1   = (float*)w;   w += MT * 512 * 4;   // [MT,512] f32
  float*  bufX2   = (float*)w;   w += MT * 512 * 4;   // [MT,512] f32
  float*  bufH    = (float*)w;                        // [MT,2048] f32

  dim3 blk(128);
  auto grd = [](int M, int N) { return dim3((unsigned)(N / 64), (unsigned)(M / 64)); };

  // ---- self attention ----
  gemm_kernel<float, __bf16><<<grd(8192, 512), blk, 0, stream>>>(tgt, sa_wq, sa_bq, nullptr, bufQ, 8192, 512, 512, 0);
  gemm_kernel<float, __bf16><<<grd(8192, 512), blk, 0, stream>>>(tgt, sa_wk, sa_bk, nullptr, bufK, 8192, 512, 512, 0);
  gemm_kernel<float, __bf16><<<grd(8192, 256), blk, 0, stream>>>(tgt, sa_wv, sa_bv, nullptr, bufV, 8192, 256, 512, 0);
  flash_attn_kernel<<<dim3(8, 8, 16), blk, 0, stream>>>(bufQ, bufK, bufV, bufAttn, 512, 512, 1);
  gemm_kernel<__bf16, float><<<grd(8192, 512), blk, 0, stream>>>(bufAttn, sa_wo, sa_bo, tgt, bufX1, 8192, 512, 256, 1);

  // ---- cross attention ----
  gemm_kernel<float, __bf16><<<grd(8192, 512),  blk, 0, stream>>>(bufX1,  ca_wq, ca_bq, nullptr, bufQ, 8192, 512, 512, 0);
  gemm_kernel<float, __bf16><<<grd(16384, 512), blk, 0, stream>>>(memory, ca_wk, ca_bk, nullptr, bufK, 16384, 512, 512, 0);
  gemm_kernel<float, __bf16><<<grd(16384, 256), blk, 0, stream>>>(memory, ca_wv, ca_bv, nullptr, bufV, 16384, 256, 512, 0);
  flash_attn_kernel<<<dim3(8, 8, 16), blk, 0, stream>>>(bufQ, bufK, bufV, bufAttn, 512, 1024, 0);
  gemm_kernel<__bf16, float><<<grd(8192, 512), blk, 0, stream>>>(bufAttn, ca_wo, ca_bo, bufX1, bufX2, 8192, 512, 256, 1);

  // ---- feed-forward (DoubleSwish fused) ----
  gemm_kernel<float, float><<<grd(8192, 2048), blk, 0, stream>>>(bufX2, ff_w1, ff_b1, nullptr, bufH, 8192, 2048, 512, 2);
  gemm_kernel<float, float><<<grd(8192, 512),  blk, 0, stream>>>(bufH,  ff_w2, ff_b2, bufX2, bufX1, 8192, 512, 2048, 1);

  // ---- BasicNorm + bypass ----
  norm_bypass_kernel<<<dim3(8192), dim3(256), 0, stream>>>(bufX1, tgt, neps, bscal, (float*)d_out);
}
